// LLaMAAttention_34574486733527
// MI455X (gfx1250) — matrix operand order repaired and checked by reading, not compiled, not run on hardware
//
#include <hip/hip_runtime.h>
#include <hip/hip_bf16.h>

typedef __bf16 bf16;
typedef __attribute__((ext_vector_type(8)))  __bf16 v8bf;
typedef __attribute__((ext_vector_type(16))) __bf16 v16bf;
typedef __attribute__((ext_vector_type(8)))  float  v8f;
typedef __attribute__((ext_vector_type(4)))  float  v4f;

#define S_LEN   2048
#define DMODEL  4096
#define KVDIM   1024
#define NHEADS  32
#define NKV     8
#define HDIM    128

__device__ __forceinline__ bf16 tobf(float f) {
    unsigned u = __builtin_bit_cast(unsigned, f);
    u += 0x7FFFu + ((u >> 16) & 1u);           // round-to-nearest-even
    unsigned short h = (unsigned short)(u >> 16);
    return __builtin_bit_cast(bf16, h);
}

__device__ __forceinline__ v8bf cvt8(v4f a, v4f b) {
    v8bf r;
    r[0]=tobf(a[0]); r[1]=tobf(a[1]); r[2]=tobf(a[2]); r[3]=tobf(a[3]);
    r[4]=tobf(b[0]); r[5]=tobf(b[1]); r[6]=tobf(b[2]); r[7]=tobf(b[3]);
    return r;
}

__device__ __forceinline__ v16bf frag2(const bf16* p0, const bf16* p1) {
    v8bf a = *(const v8bf*)p0;
    v8bf b = *(const v8bf*)p1;
    return __builtin_shufflevector(a, b, 0,1,2,3,4,5,6,7,8,9,10,11,12,13,14,15);
}

// CDNA5 async global->LDS copy, 16B per lane, tracked by ASYNCcnt.
// lds_off: LDS byte address (low 32 bits of a generic LDS pointer);
// gaddr  : 64-bit global address (VGPR pair).
__device__ __forceinline__ void async_load_b128(unsigned lds_off, const void* gptr) {
    asm volatile("global_load_async_to_lds_b128 %0, %1, off"
                 :: "v"(lds_off), "v"((unsigned long long)(size_t)gptr)
                 : "memory");
}

__device__ __forceinline__ void wait_async0() {
#if __has_builtin(__builtin_amdgcn_s_wait_asynccnt)
    __builtin_amdgcn_s_wait_asynccnt(0);
#else
    asm volatile("s_wait_asynccnt 0x0" ::: "memory");
#endif
}

// ---------------------------------------------------------------------------
// C[M,N] = A[M,K] * B[N,K]^T   (A fp32 or bf16; B fp32; C bf16 or fp32)
// block = 256 threads = 8 waves; block tile 128(M) x 128(N); K-step 32.
// Wave w computes M rows [16w,16w+16) x all 128 N cols (8 accumulators).
// ---------------------------------------------------------------------------
template<bool A_IS_F32, bool OUT_BF16>
__global__ __launch_bounds__(256)
void gemm_bt(const void* __restrict__ Av, const float* __restrict__ B,
             void* __restrict__ Cv, int M, int N, int K) {
    __shared__ bf16 As[128][40];   // 32 K padded to 40 (80B rows, 16B aligned)
    __shared__ bf16 Bs[128][40];

    const int tid  = threadIdx.x;
    const int wave = tid >> 5;
    const int lane = tid & 31;
    const int half = lane >> 4;      // lane-half selects K sub-range
    const int ln   = lane & 15;
    const int mblk = blockIdx.y * 128;
    const int nblk = blockIdx.x * 128;

    const int lrow = tid >> 1;           // 0..127
    const int lcol = (tid & 1) * 16;     // 0 or 16

    v8f acc[8] = {};

    for (int k0 = 0; k0 < K; k0 += 32) {
        // ---- stage A tile (128x32) into LDS as bf16 ----
        if (A_IS_F32) {
            const float* src = (const float*)Av + (size_t)(mblk + lrow) * K + k0 + lcol;
            v4f a0 = ((const v4f*)src)[0], a1 = ((const v4f*)src)[1];
            v4f a2 = ((const v4f*)src)[2], a3 = ((const v4f*)src)[3];
            *(v8bf*)&As[lrow][lcol]     = cvt8(a0, a1);
            *(v8bf*)&As[lrow][lcol + 8] = cvt8(a2, a3);
        } else {
            const bf16* src = (const bf16*)Av + (size_t)(mblk + lrow) * K + k0 + lcol;
            *(v8bf*)&As[lrow][lcol]     = ((const v8bf*)src)[0];
            *(v8bf*)&As[lrow][lcol + 8] = ((const v8bf*)src)[1];
        }
        // ---- stage B tile (128x32) ----
        {
            const float* src = B + (size_t)(nblk + lrow) * K + k0 + lcol;
            __builtin_prefetch(src + 32, 0, 0);      // next K-tile -> global_prefetch
            v4f b0 = ((const v4f*)src)[0], b1 = ((const v4f*)src)[1];
            v4f b2 = ((const v4f*)src)[2], b3 = ((const v4f*)src)[3];
            *(v8bf*)&Bs[lrow][lcol]     = cvt8(b0, b1);
            *(v8bf*)&Bs[lrow][lcol + 8] = cvt8(b2, b3);
        }
        __syncthreads();

        // A fragment: lanes 0-15 hold K{0..7,16..23}; lanes 16-31 K{8..15,24..31}
        const bf16* ap = &As[wave * 16 + ln][half ? 8 : 0];
        v16bf afrag = frag2(ap, ap + 16);
        #pragma unroll
        for (int nt = 0; nt < 8; ++nt) {
            // B fragment: lanes 0-15 hold K{0..7,16..23}; lanes 16-31 K{8..15,24..31}
            const bf16* bp = &Bs[nt * 16 + ln][half * 8];
            v16bf bfrag = frag2(bp, bp + 16);
            acc[nt] = __builtin_amdgcn_wmma_f32_16x16x32_bf16(
                false, afrag, false, bfrag, (short)0, acc[nt], false, false);
        }
        __syncthreads();
    }

    #pragma unroll
    for (int nt = 0; nt < 8; ++nt) {
        #pragma unroll
        for (int r = 0; r < 8; ++r) {
            int m = mblk + wave * 16 + r + half * 8;   // C layout: VGPR r -> M
            int n = nblk + nt * 16 + ln;               // lane -> N
            float v = acc[nt][r];
            if (OUT_BF16) ((bf16*)Cv)[(size_t)m * N + n] = tobf(v);
            else          ((float*)Cv)[(size_t)m * N + n] = v;
        }
    }
}

// ---------------------------------------------------------------------------
// RoPE in-place on bf16 [S, ncols]; ncols = heads*128; pairs (i, i+64).
// ---------------------------------------------------------------------------
__global__ void rope_kernel(bf16* __restrict__ buf, int S, int ncols) {
    int idx = blockIdx.x * blockDim.x + threadIdx.x;
    int halfcols = ncols >> 1;
    if (idx >= S * halfcols) return;
    int s = idx / halfcols;
    int c = idx - s * halfcols;
    int head = c >> 6;
    int i = c & 63;
    size_t base = (size_t)s * ncols + head * HDIM;
    float x1 = (float)buf[base + i];
    float x2 = (float)buf[base + i + 64];
    // freq = theta^(-2i/128); ln(10000)/128 = 0.0719558...
    float freq = __expf(-0.14391156f * (float)i);
    float ang = (float)s * freq;
    float sn, cs;
    __sincosf(ang, &sn, &cs);
    buf[base + i]      = tobf(x1 * cs - x2 * sn);
    buf[base + i + 64] = tobf(x2 * cs + x1 * sn);
}

// ---------------------------------------------------------------------------
// Flash attention: block = (head, 64-query tile), 4 waves x 16 query rows.
// K tile staged with async global->LDS copies; V transposed for PV B-operand.
// ---------------------------------------------------------------------------
__global__ __launch_bounds__(128)
void attn_kernel(const bf16* __restrict__ Qb, const bf16* __restrict__ Kb,
                 const bf16* __restrict__ Vb, bf16* __restrict__ Ctx) {
    __shared__ bf16 Ks[64][136];       // [key][d]   (pad -> 272B rows)
    __shared__ bf16 Vt[128][72];       // [d][key]   (pad -> 144B rows)
    __shared__ bf16 Ps[4][16][72];     // per-wave P repack buffer

    const int h   = blockIdx.y;
    const int kvh = h >> 2;                     // GQA: 4 Q heads per KV head
    const int qb  = blockIdx.x * 64;
    const int tid  = threadIdx.x;
    const int wave = tid >> 5;
    const int lane = tid & 31;
    const int half = lane >> 4;
    const int ln   = lane & 15;
    const int q0   = qb + wave * 16;

    // Q fragments for this wave's 16 rows (direct from global, b128 loads)
    v16bf aq[4];
    {
        const bf16* qrow = Qb + (size_t)(q0 + ln) * DMODEL + h * HDIM;
        #pragma unroll
        for (int ks = 0; ks < 4; ++ks) {
            const bf16* p = qrow + ks * 32 + (half ? 8 : 0);
            aq[ks] = frag2(p, p + 16);
        }
    }

    float mrow[8], lrow[8];
    v8f ob[8] = {};
    #pragma unroll
    for (int r = 0; r < 8; ++r) { mrow[r] = -3.0e38f; lrow[r] = 0.0f; }

    const float scale = 0.08838834764831845f;   // 1/sqrt(128)
    const int krow  = tid >> 1;
    const int dhalf = (tid & 1) * 64;

    for (int kb = 0; kb <= qb; kb += 64) {
        // ---- K tile: async global->LDS (ASYNCcnt), no VGPR round-trip ----
        {
            const bf16* src = Kb + (size_t)(kb + krow) * KVDIM + kvh * HDIM + dhalf;
            unsigned ldsb = (unsigned)(size_t)(void*)&Ks[krow][dhalf];
            #pragma unroll
            for (int j = 0; j < 8; ++j)
                async_load_b128(ldsb + 16u * j, src + 8 * j);
        }
        // ---- V tile, transposed into Vt[d][key] (overlaps with async K) ----
        {
            const bf16* src = Vb + (size_t)(kb + krow) * KVDIM + kvh * HDIM + dhalf;
            #pragma unroll
            for (int j0 = 0; j0 < 64; j0 += 8) {
                v8bf chunk = *(const v8bf*)(src + j0);
                #pragma unroll
                for (int j = 0; j < 8; ++j)
                    Vt[dhalf + j0 + j][krow] = chunk[j];
            }
        }
        wait_async0();          // our async copies are invisible to the compiler
        __syncthreads();

        // ---- S = Q K^T over d=128 (4 WMMA k-steps per 16x16 tile) ----
        v8f sf[4] = {};
        #pragma unroll
        for (int nt = 0; nt < 4; ++nt) {
            #pragma unroll
            for (int ks = 0; ks < 4; ++ks) {
                const bf16* bp = &Ks[nt * 16 + ln][ks * 32 + half * 8];
                v16bf bfrag = frag2(bp, bp + 16);
                sf[nt] = __builtin_amdgcn_wmma_f32_16x16x32_bf16(
                    false, aq[ks], false, bfrag, (short)0, sf[nt], false, false);
            }
        }

        // ---- scale + causal mask + block row-max ----
        float bmax[8];
        #pragma unroll
        for (int r = 0; r < 8; ++r) bmax[r] = -3.0e38f;
        #pragma unroll
        for (int nt = 0; nt < 4; ++nt) {
            #pragma unroll
            for (int r = 0; r < 8; ++r) {
                float sc = sf[nt][r] * scale;
                if (kb + nt * 16 + ln > q0 + r + half * 8) sc = -1.0e30f;
                sf[nt][r] = sc;
                bmax[r] = fmaxf(bmax[r], sc);
            }
        }
        #pragma unroll
        for (int r = 0; r < 8; ++r) {
            float v = bmax[r];
            v = fmaxf(v, __shfl_xor(v, 1, 32));
            v = fmaxf(v, __shfl_xor(v, 2, 32));
            v = fmaxf(v, __shfl_xor(v, 4, 32));
            v = fmaxf(v, __shfl_xor(v, 8, 32));
            bmax[r] = v;
        }

        // ---- online softmax update ----
        float alpha[8], rsum[8];
        #pragma unroll
        for (int r = 0; r < 8; ++r) {
            float nm = fmaxf(mrow[r], bmax[r]);
            alpha[r] = __expf(mrow[r] - nm);
            mrow[r] = nm;
            rsum[r] = 0.0f;
        }
        #pragma unroll
        for (int nt = 0; nt < 4; ++nt) {
            #pragma unroll
            for (int r = 0; r < 8; ++r) {
                float p = __expf(sf[nt][r] - mrow[r]);
                rsum[r] += p;
                Ps[wave][r + half * 8][nt * 16 + ln] = tobf(p);  // repack C->A
            }
        }
        #pragma unroll
        for (int r = 0; r < 8; ++r) {
            float v = rsum[r];
            v += __shfl_xor(v, 1, 32);
            v += __shfl_xor(v, 2, 32);
            v += __shfl_xor(v, 4, 32);
            v += __shfl_xor(v, 8, 32);
            lrow[r] = lrow[r] * alpha[r] + v;
        }
        #pragma unroll
        for (int dt = 0; dt < 8; ++dt)
            #pragma unroll
            for (int r = 0; r < 8; ++r)
                ob[dt][r] *= alpha[r];

        // ---- O += P V  (k=64 keys -> 2 WMMA k-steps, 8 d-tiles) ----
        #pragma unroll
        for (int ks2 = 0; ks2 < 2; ++ks2) {
            const bf16* pp = &Ps[wave][ln][ks2 * 32 + (half ? 8 : 0)];
            v16bf pf = frag2(pp, pp + 16);
            #pragma unroll
            for (int dt = 0; dt < 8; ++dt) {
                const bf16* vp = &Vt[dt * 16 + ln][ks2 * 32 + half * 8];
                v16bf vf = frag2(vp, vp + 16);
                ob[dt] = __builtin_amdgcn_wmma_f32_16x16x32_bf16(
                    false, pf, false, vf, (short)0, ob[dt], false, false);
            }
        }
        __syncthreads();
    }

    // ---- normalize and write context [s, h*128 + d] ----
    #pragma unroll
    for (int dt = 0; dt < 8; ++dt) {
        #pragma unroll
        for (int r = 0; r < 8; ++r) {
            int m = q0 + r + half * 8;
            float v = ob[dt][r] / lrow[r];
            Ctx[(size_t)m * DMODEL + h * HDIM + dt * 16 + ln] = tobf(v);
        }
    }
}

// ---------------------------------------------------------------------------
extern "C" void kernel_launch(void* const* d_in, const int* in_sizes, int n_in,
                              void* d_out, int out_size, void* d_ws, size_t ws_size,
                              hipStream_t stream) {
    (void)in_sizes; (void)n_in; (void)out_size; (void)ws_size;
    const float* X  = (const float*)d_in[0];
    const float* Wq = (const float*)d_in[1];
    const float* Wk = (const float*)d_in[2];
    const float* Wv = (const float*)d_in[3];
    const float* Wo = (const float*)d_in[4];
    float* out = (float*)d_out;

    bf16* Qb  = (bf16*)d_ws;
    bf16* Kb  = Qb + (size_t)S_LEN * DMODEL;
    bf16* Vb  = Kb + (size_t)S_LEN * KVDIM;
    bf16* Ctx = Vb + (size_t)S_LEN * KVDIM;

    dim3 blk(256);
    // Q/K/V projections (C = X * W^T), bf16 outputs
    gemm_bt<true,  true ><<<dim3(DMODEL/128, S_LEN/128), blk, 0, stream>>>(X, Wq, Qb, S_LEN, DMODEL, DMODEL);
    gemm_bt<true,  true ><<<dim3(KVDIM /128, S_LEN/128), blk, 0, stream>>>(X, Wk, Kb, S_LEN, KVDIM,  DMODEL);
    gemm_bt<true,  true ><<<dim3(KVDIM /128, S_LEN/128), blk, 0, stream>>>(X, Wv, Vb, S_LEN, KVDIM,  DMODEL);
    // RoPE on Q and K
    rope_kernel<<<(S_LEN * (DMODEL/2) + 255) / 256, 256, 0, stream>>>(Qb, S_LEN, DMODEL);
    rope_kernel<<<(S_LEN * (KVDIM /2) + 255) / 256, 256, 0, stream>>>(Kb, S_LEN, KVDIM);
    // causal flash attention, GQA 4:1
    attn_kernel<<<dim3(S_LEN/64, NHEADS), dim3(128), 0, stream>>>(Qb, Kb, Vb, Ctx);
    // output projection -> fp32
    gemm_bt<false, false><<<dim3(DMODEL/128, S_LEN/128), blk, 0, stream>>>(Ctx, Wo, out, S_LEN, DMODEL, DMODEL);
}
